// GraphFormers_24129126269639
// MI455X (gfx1250) — matrix operand order repaired and checked by reading, not compiled, not run on hardware
//
#include <hip/hip_runtime.h>

typedef unsigned short u16;
typedef unsigned int   u32;
typedef __attribute__((ext_vector_type(16))) __bf16 v16bf;
typedef __attribute__((ext_vector_type(8)))  float  v8f;
typedef __attribute__((ext_vector_type(4)))  int    v4i;

#define AN      512          // B*N sequences
#define SEQ     65           // L+1
#define LTOK    64
#define DMODEL  512
#define NHEAD   8
#define DHEAD   64
#define FFDIM   2048
#define NROWS   (AN*SEQ)     // 33280
#define QKVN    1536

#if defined(__has_builtin)
#if __has_builtin(__builtin_amdgcn_global_load_async_to_lds_b128) && \
    __has_builtin(__builtin_amdgcn_s_wait_asynccnt)
#define HAVE_ASYNC_LDS 1
#endif
#endif
#ifndef HAVE_ASYNC_LDS
#define HAVE_ASYNC_LDS 0
#endif

#if HAVE_ASYNC_LDS
typedef __attribute__((address_space(1))) v4i* gv4i_p;
typedef __attribute__((address_space(3))) v4i* lv4i_p;
#endif

// ---------------- bf16 helpers ----------------
__device__ __forceinline__ u16 f2bf(float f) {
  u32 u = __float_as_uint(f);
  u = (u + 0x7FFFu + ((u >> 16) & 1u)) >> 16;
  return (u16)u;
}
__device__ __forceinline__ float bf2f(u16 h) {
  return __uint_as_float(((u32)h) << 16);
}

struct __align__(8) us4 { u16 x, y, z, w; };

union FragU { uint4 q[2]; v16bf v; };

// A fragment 16x32 bf16: lane<16 holds row M=l, K {c..c+7, c+16..c+23}, c = k0 (+8 for hi lanes)
__device__ __forceinline__ v16bf ldsA_frag(const u16* base, int row0, int stride, int k0) {
  int lane = threadIdx.x & 31; int l = lane & 15; int hi = (lane >> 4) & 1;
  const u16* p = base + (size_t)(row0 + l) * stride + k0 + (hi ? 8 : 0);
  FragU f; f.q[0] = *(const uint4*)p; f.q[1] = *(const uint4*)(p + 16);
  return f.v;
}
// B fragment 32x16 bf16 from N-major LDS (ldsB[n][k]): lane<16 col N=l, K {c..c+7, c+16..c+23}, c = k0 (+8 for hi lanes)
__device__ __forceinline__ v16bf ldsB_frag(const u16* base, int n0, int stride, int k0) {
  int lane = threadIdx.x & 31; int l = lane & 15; int hi = (lane >> 4) & 1;
  const u16* p = base + (size_t)(n0 + l) * stride + k0 + (hi ? 8 : 0);
  FragU f; f.q[0] = *(const uint4*)p; f.q[1] = *(const uint4*)(p + 16);
  return f.v;
}

#define WMMA_BF16(A, B, C) \
  __builtin_amdgcn_wmma_f32_16x16x32_bf16(false, (A), false, (B), (short)0, (C), false, false)

// Copy one 16-byte chunk global -> LDS (async on CDNA5, fallback via VGPRs)
__device__ __forceinline__ void g2lds_b128(const u16* gsrc, u16* ldst) {
#if HAVE_ASYNC_LDS
  __builtin_amdgcn_global_load_async_to_lds_b128((gv4i_p)gsrc, (lv4i_p)ldst, 0, 0);
#else
  *(uint4*)ldst = *(const uint4*)gsrc;
#endif
}
__device__ __forceinline__ void g2lds_fence() {
#if HAVE_ASYNC_LDS
  __builtin_amdgcn_s_wait_asynccnt(0);
#endif
}

// ---------------- GEMM: C(MxN) = A(MxK) @ Wt(NxK)^T + bias ----------------
// Block tile 128x128, 8 waves, each wave 32x64 (2 A-frags x 4 B-frags = 8 WMMA / K-step).
// EPI 0: bf16 store; 1: bf16 gelu; 2: fp32 += residual (in-place on outF)
template <int EPI>
__global__ __launch_bounds__(256) void gemm_bf16(
    const u16* __restrict__ A, const u16* __restrict__ Bt,
    const float* __restrict__ bias, u16* __restrict__ outB,
    float* __restrict__ outF, int M, int N, int K)
{
  __shared__ __align__(16) u16 sA[128 * 32];
  __shared__ __align__(16) u16 sB[128 * 32];
  int mb = blockIdx.x, nb = blockIdx.y;
  int tid = threadIdx.x, wave = tid >> 5, lane = tid & 31;
  int wm = wave >> 1, wn = wave & 1;

  v8f acc[2][4] = { { {0,0,0,0,0,0,0,0}, {0,0,0,0,0,0,0,0},
                      {0,0,0,0,0,0,0,0}, {0,0,0,0,0,0,0,0} },
                    { {0,0,0,0,0,0,0,0}, {0,0,0,0,0,0,0,0},
                      {0,0,0,0,0,0,0,0}, {0,0,0,0,0,0,0,0} } };

  const u16* Ab = A  + (size_t)(mb * 128) * K;
  const u16* Bb = Bt + (size_t)(nb * 128) * K;

  for (int k0 = 0; k0 < K; k0 += 32) {
    __syncthreads();
    // A tile 128x32 and B tile 128x32 (N-major): 512 chunks each, 2 per thread
    #pragma unroll
    for (int i = 0; i < 2; i++) {
      int ch = tid * 2 + i; int r = ch >> 2, c = (ch & 3) * 8;
      g2lds_b128(&Ab[(size_t)r * K + k0 + c], &sA[r * 32 + c]);
      g2lds_b128(&Bb[(size_t)r * K + k0 + c], &sB[r * 32 + c]);
    }
    g2lds_fence();
    __syncthreads();
    v16bf a0 = ldsA_frag(sA, wm * 32,      32, 0);
    v16bf a1 = ldsA_frag(sA, wm * 32 + 16, 32, 0);
    #pragma unroll
    for (int t = 0; t < 4; t++) {
      v16bf b = ldsB_frag(sB, wn * 64 + t * 16, 32, 0);
      acc[0][t] = WMMA_BF16(a0, b, acc[0][t]);
      acc[1][t] = WMMA_BF16(a1, b, acc[1][t]);
    }
  }

  int l = lane & 15, hi = (lane >> 4) & 1;
  #pragma unroll
  for (int t = 0; t < 4; t++) {
    int c = nb * 128 + wn * 64 + t * 16 + l;
    float bv = bias ? bias[c] : 0.0f;
    #pragma unroll
    for (int mi = 0; mi < 2; mi++) {
      #pragma unroll
      for (int r = 0; r < 8; r++) {
        int m = mb * 128 + wm * 32 + mi * 16 + hi * 8 + r;
        float v = acc[mi][t][r] + bv;
        size_t idx = (size_t)m * N + c;
        if (EPI == 0) {
          outB[idx] = f2bf(v);
        } else if (EPI == 1) {
          outB[idx] = f2bf(0.5f * v * (1.0f + erff(v * 0.70710678f)));
        } else {
          outF[idx] = v + outF[idx];
        }
      }
    }
  }
}

// ---------------- fused attention per (sequence, head) ----------------
__global__ __launch_bounds__(256) void attn_kernel(
    const u16* __restrict__ qkv, const float* __restrict__ relb,
    const float* __restrict__ peak, u16* __restrict__ ctx)
{
  int an = blockIdx.x, h = blockIdx.y;
  __shared__ __align__(16) char smem[58368];
  u16*   sQ  = (u16*)smem;                       // 80x64
  u16*   sK  = sQ + 80 * 64;                     // 80x64
  float* sS  = (float*)(smem + 20480);           // 80x80 f32
  u16*   sVt = (u16*)(smem + 20480 + 25600);     // 64x96 (dim-major)
  u16*   sP  = (u16*)smem;                       // 80x96 (reuses sQ/sK)

  int tid = threadIdx.x, wave = tid >> 5, lane = tid & 31;
  int l = lane & 15, hig = (lane >> 4) & 1;
  const size_t rowbase = (size_t)an * SEQ * QKVN;

  for (int idx = tid; idx < SEQ * 64; idx += 256) {
    int s = idx >> 6, d = idx & 63;
    size_t g = rowbase + (size_t)s * QKVN + h * 64 + d;
    sQ[s * 64 + d] = qkv[g];
    sK[s * 64 + d] = qkv[g + 512];
  }
  for (int idx = tid; idx < 64 * 96; idx += 256) {
    int d = idx / 96, s = idx % 96;
    sVt[idx] = (s < SEQ) ? qkv[rowbase + (size_t)s * QKVN + 1024 + h * 64 + d] : (u16)0;
  }
  __syncthreads();

  // raw scores = Q K^T / 8   (5x5 tiles of 16, K=64); bias added in softmax pass
  for (int t = wave; t < 25; t += 8) {
    int mt = t / 5, nt = t % 5;
    v8f acc = {0,0,0,0,0,0,0,0};
    {
      v16bf a = ldsA_frag(sQ, mt * 16, 64, 0);
      v16bf b = ldsB_frag(sK, nt * 16, 64, 0);
      acc = WMMA_BF16(a, b, acc);
    }
    {
      v16bf a = ldsA_frag(sQ, mt * 16, 64, 32);
      v16bf b = ldsB_frag(sK, nt * 16, 64, 32);
      acc = WMMA_BF16(a, b, acc);
    }
    int n = nt * 16 + l;
    #pragma unroll
    for (int r = 0; r < 8; r++)
      sS[(mt * 16 + hig * 8 + r) * 80 + n] = acc[r] * 0.125f;
  }
  __syncthreads();

  // softmax, wave per row, bias added here (coalesced: key index == lane offset)
  const float* pk = peak + an * SEQ;
  for (int r = wave; r < SEQ; r += 8) {
    const float* rb = relb + ((size_t)h * SEQ + r) * SEQ;
    float s0 = sS[r * 80 + lane]      + pk[lane]      + rb[lane];        // k = 0..31
    float s1 = sS[r * 80 + 32 + lane] + pk[32 + lane] + rb[32 + lane];   // k = 32..63
    float s2 = (lane == 0) ? (sS[r * 80 + 64] + pk[64] + rb[64]) : -3.0e38f; // k = 64
    float mx = fmaxf(fmaxf(s0, s1), s2);
    #pragma unroll
    for (int o = 16; o; o >>= 1) mx = fmaxf(mx, __shfl_xor(mx, o, 32));
    float e0 = __expf(s0 - mx), e1 = __expf(s1 - mx);
    float e2 = (lane == 0) ? __expf(s2 - mx) : 0.0f;
    float sm = e0 + e1 + e2;
    #pragma unroll
    for (int o = 16; o; o >>= 1) sm += __shfl_xor(sm, o, 32);
    float inv = 1.0f / sm;
    sP[r * 96 + lane]      = f2bf(e0 * inv);
    sP[r * 96 + 32 + lane] = f2bf(e1 * inv);
    if (lane == 0) sP[r * 96 + 64] = f2bf(e2 * inv);
    if (lane < 31) sP[r * 96 + 65 + lane] = 0;
  }
  for (int r = SEQ + wave; r < 80; r += 8) {
    sP[r * 96 + lane] = 0; sP[r * 96 + 32 + lane] = 0; sP[r * 96 + 64 + lane] = 0;
  }
  __syncthreads();

  // O = P @ V  (5x4 tiles, K=96 in 3 steps)
  for (int t = wave; t < 20; t += 8) {
    int mt = t / 4, nt = t % 4;
    v8f acc = {0,0,0,0,0,0,0,0};
    #pragma unroll
    for (int ks = 0; ks < 3; ks++) {
      v16bf a = ldsA_frag(sP,  mt * 16, 96, ks * 32);
      v16bf b = ldsB_frag(sVt, nt * 16, 96, ks * 32);
      acc = WMMA_BF16(a, b, acc);
    }
    int d = nt * 16 + l;
    #pragma unroll
    for (int r = 0; r < 8; r++) {
      int m = mt * 16 + hig * 8 + r;
      if (m < SEQ)
        ctx[((size_t)(an * SEQ + m)) * DMODEL + h * 64 + d] = f2bf(acc[r]);
    }
  }
}

// ---------------- LayerNorm (in-place fp32 + bf16 copy), 128 thr/row ----------------
__device__ __forceinline__ void ln_core(float4 v, const float* g, const float* b,
                                        float* p, u16* q, int t,
                                        float* r1, float* r2) {
  float s1 = v.x + v.y + v.z + v.w;
  float s2 = v.x * v.x + v.y * v.y + v.z * v.z + v.w * v.w;
  #pragma unroll
  for (int o = 16; o; o >>= 1) { s1 += __shfl_xor(s1, o, 32); s2 += __shfl_xor(s2, o, 32); }
  int wave = t >> 5;
  if ((t & 31) == 0) { r1[wave] = s1; r2[wave] = s2; }
  __syncthreads();
  s1 = r1[0] + r1[1] + r1[2] + r1[3];
  s2 = r2[0] + r2[1] + r2[2] + r2[3];
  float mu = s1 * (1.0f / DMODEL);
  float var = s2 * (1.0f / DMODEL) - mu * mu;
  float inv = rsqrtf(var + 1e-6f);
  float4 gv = ((const float4*)g)[t], bv = ((const float4*)b)[t];
  float o0 = (v.x - mu) * inv * gv.x + bv.x;
  float o1 = (v.y - mu) * inv * gv.y + bv.y;
  float o2 = (v.z - mu) * inv * gv.z + bv.z;
  float o3 = (v.w - mu) * inv * gv.w + bv.w;
  ((float4*)p)[t] = make_float4(o0, o1, o2, o3);
  us4 ub; ub.x = f2bf(o0); ub.y = f2bf(o1); ub.z = f2bf(o2); ub.w = f2bf(o3);
  *(us4*)(q + t * 4) = ub;
}

__global__ __launch_bounds__(128) void ln_kernel(float* __restrict__ x,
    const float* __restrict__ g, const float* __restrict__ b, u16* __restrict__ xb)
{
  __shared__ float r1[4], r2[4];
  size_t row = blockIdx.x; int t = threadIdx.x;
  float* p = x + row * DMODEL;
  float4 v = ((const float4*)p)[t];
  ln_core(v, g, b, p, xb + row * DMODEL, t, r1, r2);
}

// ---------------- embedding gather + LN ----------------
__global__ __launch_bounds__(128) void embed_kernel(const int* __restrict__ ids,
    const float* __restrict__ we, const float* __restrict__ pe,
    const float* __restrict__ g, const float* __restrict__ b,
    float* __restrict__ hidden, u16* __restrict__ xb)
{
  __shared__ float r1[4], r2[4];
  size_t row = blockIdx.x; int t = threadIdx.x;
  int an = (int)(row / SEQ), s = (int)(row % SEQ);
  float* p = hidden + row * DMODEL;
  u16*   q = xb + row * DMODEL;
  if (s == 0) {
    ((float4*)p)[t] = make_float4(0, 0, 0, 0);
    us4 z = {0, 0, 0, 0}; *(us4*)(q + t * 4) = z;
    return;
  }
  int id = ids[an * LTOK + (s - 1)];
  float4 v  = ((const float4*)(we + (size_t)id * DMODEL))[t];
  float4 pv = ((const float4*)(pe + (size_t)(s - 1) * DMODEL))[t];
  v.x += pv.x; v.y += pv.y; v.z += pv.z; v.w += pv.w;
  ln_core(v, g, b, p, q, t, r1, r2);
}

// ---------------- small setup kernels ----------------
__global__ void nodemask_kernel(const int* __restrict__ ids, float* __restrict__ nm) {
  int i = blockIdx.x * 256 + threadIdx.x;
  if (i >= AN) return;
  const int* p = ids + (size_t)i * LTOK;
  int any = 0;
  for (int l = 0; l < LTOK; l++) any |= p[l];
  nm[i] = any ? 1.0f : 0.0f;
}

__global__ void relb_kernel(const float* __restrict__ relw, float* __restrict__ relb) {
  int i = blockIdx.x * 256 + threadIdx.x;
  if (i >= NHEAD * SEQ * SEQ) return;
  int h = i / (SEQ * SEQ), r = i % (SEQ * SEQ);
  int qi = r / SEQ, ki = r % SEQ;
  int bucket;
  if (ki == 0) bucket = 32;                 // BINS
  else if (qi == 0) bucket = 0;
  else {
    int rp = ki - qi;
    int ret = (rp > 0) ? 16 : 0;
    int n = rp < 0 ? -rp : rp;
    int vl = 8 + (int)(logf(fmaxf((float)n, 1.0f) / 8.0f) / logf(16.0f) * 8.0f);
    vl = vl < 15 ? vl : 15;
    bucket = ret + (n < 8 ? n : vl);
  }
  relb[i] = relw[bucket * NHEAD + h];
}

__global__ void peak_kernel(const float* __restrict__ amask, float* __restrict__ peak) {
  int i = blockIdx.x * 256 + threadIdx.x;
  if (i >= AN * SEQ) return;
  int an = i / SEQ, s = i % SEQ;
  peak[i] = (s == 0) ? 0.0f : (1.0f - amask[an * LTOK + s - 1]) * -10000.0f;
}

__global__ void nodeadd_kernel(const float* __restrict__ nm, const float* __restrict__ relw,
                               float* __restrict__ na) {
  int i = blockIdx.x * 256 + threadIdx.x;
  if (i >= 16 * NHEAD * 32) return;
  int b = i / (NHEAD * 32), r = i % (NHEAD * 32);
  int h = r / 32, n = r % 32;
  na[i] = (1.0f - nm[b * 32 + n]) * -10000.0f + relw[(n == 0 ? 0 : 33) * NHEAD + h];
}

__global__ void transpose_bf16_kernel(const float* __restrict__ src, u16* __restrict__ dst,
                                      int K, int N) {
  int i = blockIdx.x * 256 + threadIdx.x;
  if (i >= K * N) return;
  int k = i / N, n = i % N;
  dst[(size_t)n * K + k] = f2bf(src[i]);
}

__global__ void concat3_kernel(const float* __restrict__ a, const float* __restrict__ b,
                               const float* __restrict__ c, float* __restrict__ dst) {
  int i = blockIdx.x * 256 + threadIdx.x;
  if (i >= 3 * DMODEL) return;
  dst[i] = (i < DMODEL) ? a[i] : (i < 2 * DMODEL) ? b[i - DMODEL] : c[i - 2 * DMODEL];
}

__global__ void gather_cls_kernel(const float* __restrict__ hidden, u16* __restrict__ xg) {
  int i = blockIdx.x * 256 + threadIdx.x;
  if (i >= AN * DMODEL) return;
  int row = i / DMODEL, d = i % DMODEL;
  xg[i] = f2bf(hidden[((size_t)row * SEQ + 1) * DMODEL + d]);
}

// station = MHA over the 32 nodes of each graph (no out-proj)
__global__ __launch_bounds__(32) void graph_attn_kernel(
    const u16* __restrict__ qkvg, const float* __restrict__ na,
    float* __restrict__ hidden, u16* __restrict__ xb)
{
  __shared__ float sc[32 * 33];
  int b = blockIdx.x, h = blockIdx.y, qn = threadIdx.x;
  float q[DHEAD];
  const u16* qrow = qkvg + (size_t)(b * 32 + qn) * QKVN + h * 64;
  #pragma unroll
  for (int d = 0; d < 64; d++) q[d] = bf2f(qrow[d]);
  for (int kn = 0; kn < 32; kn++) {
    const u16* krow = qkvg + (size_t)(b * 32 + kn) * QKVN + 512 + h * 64;
    float s = 0.0f;
    #pragma unroll
    for (int d = 0; d < 64; d++) s += q[d] * bf2f(krow[d]);
    sc[qn * 33 + kn] = s * 0.125f + na[(b * NHEAD + h) * 32 + kn];
  }
  float mx = -3.0e38f;
  for (int kn = 0; kn < 32; kn++) mx = fmaxf(mx, sc[qn * 33 + kn]);
  float sum = 0.0f;
  for (int kn = 0; kn < 32; kn++) { float e = __expf(sc[qn * 33 + kn] - mx); sc[qn * 33 + kn] = e; sum += e; }
  float inv = 1.0f / sum;
  size_t orow = ((size_t)(b * 32 + qn) * SEQ) * DMODEL + h * 64;
  for (int d = 0; d < 64; d++) {
    float o = 0.0f;
    for (int kn = 0; kn < 32; kn++)
      o += sc[qn * 33 + kn] * bf2f(qkvg[(size_t)(b * 32 + kn) * QKVN + 1024 + h * 64 + d]);
    o *= inv;
    hidden[orow + d] = o;
    xb[orow + d] = f2bf(o);
  }
}

// ---------------- host ----------------
extern "C" void kernel_launch(void* const* d_in, const int* in_sizes, int n_in,
                              void* d_out, int out_size, void* d_ws, size_t ws_size,
                              hipStream_t stream) {
  (void)in_sizes; (void)n_in; (void)out_size; (void)ws_size;
  const int*   ids   = (const int*)  d_in[0];
  const float* amask = (const float*)d_in[1];
  const float* wemb  = (const float*)d_in[2];
  const float* pemb  = (const float*)d_in[3];
  const float* embg  = (const float*)d_in[4];
  const float* embb  = (const float*)d_in[5];
  const float* relw  = (const float*)d_in[6];
  const float* g_qw = (const float*)d_in[7],  *g_qb = (const float*)d_in[8];
  const float* g_kw = (const float*)d_in[9],  *g_kb = (const float*)d_in[10];
  const float* g_vw = (const float*)d_in[11], *g_vb = (const float*)d_in[12];
  auto LP = [&](int i, int j) { return (const float*)d_in[13 + 16 * i + j]; };
  // j: 0 qw 1 qb 2 kw 3 kb 4 vw 5 vb 6 ow 7 ob 8 ln1g 9 ln1b 10 fw 11 fb 12 f2w 13 f2b 14 ln2g 15 ln2b

  float* hidden = (float*)d_out;                    // (NROWS, 512) fp32, final answer
  float* nmOut  = hidden + (size_t)NROWS * DMODEL;  // (512,)

  char* p = (char*)d_ws;
  auto alloc = [&](size_t bytes) { char* r = p; p += (bytes + 255) & ~(size_t)255; return r; };
  float* relb   = (float*)alloc((size_t)NHEAD * SEQ * SEQ * 4);
  float* peak   = (float*)alloc((size_t)AN * SEQ * 4);
  float* nodea  = (float*)alloc((size_t)16 * NHEAD * 32 * 4);
  float* qkvbL  = (float*)alloc((size_t)4 * QKVN * 4);
  float* gqkvb  = (float*)alloc((size_t)QKVN * 4);
  u16* xb   = (u16*)alloc((size_t)NROWS * DMODEL * 2);
  u16* qkv  = (u16*)alloc((size_t)NROWS * QKVN * 2);
  u16* ctx  = (u16*)alloc((size_t)NROWS * DMODEL * 2);
  u16* hbuf = (u16*)alloc((size_t)NROWS * FFDIM * 2);
  u16* xg   = (u16*)alloc((size_t)AN * DMODEL * 2);
  u16* qkvg = (u16*)alloc((size_t)AN * QKVN * 2);
  u16* wqkv = (u16*)alloc((size_t)4 * QKVN * DMODEL * 2);
  u16* wo   = (u16*)alloc((size_t)4 * DMODEL * DMODEL * 2);
  u16* wf   = (u16*)alloc((size_t)4 * FFDIM * DMODEL * 2);
  u16* wf2  = (u16*)alloc((size_t)4 * DMODEL * FFDIM * 2);
  u16* wg   = (u16*)alloc((size_t)QKVN * DMODEL * 2);

  auto T = [&](const float* src, u16* dst, int K, int N) {
    int total = K * N;
    transpose_bf16_kernel<<<(total + 255) / 256, 256, 0, stream>>>(src, dst, K, N);
  };

  // setup
  nodemask_kernel<<<(AN + 255) / 256, 256, 0, stream>>>(ids, nmOut);
  relb_kernel<<<(NHEAD * SEQ * SEQ + 255) / 256, 256, 0, stream>>>(relw, relb);
  peak_kernel<<<(AN * SEQ + 255) / 256, 256, 0, stream>>>(amask, peak);
  nodeadd_kernel<<<(16 * NHEAD * 32 + 255) / 256, 256, 0, stream>>>(nmOut, relw, nodea);

  for (int i = 0; i < 4; i++) {
    u16* wq = wqkv + (size_t)i * QKVN * DMODEL;
    T(LP(i, 0), wq, DMODEL, DMODEL);
    T(LP(i, 2), wq + (size_t)DMODEL * DMODEL, DMODEL, DMODEL);
    T(LP(i, 4), wq + (size_t)2 * DMODEL * DMODEL, DMODEL, DMODEL);
    T(LP(i, 6), wo + (size_t)i * DMODEL * DMODEL, DMODEL, DMODEL);
    T(LP(i, 10), wf + (size_t)i * FFDIM * DMODEL, DMODEL, FFDIM);
    T(LP(i, 12), wf2 + (size_t)i * DMODEL * FFDIM, FFDIM, DMODEL);
    concat3_kernel<<<(QKVN + 255) / 256, 256, 0, stream>>>(LP(i, 1), LP(i, 3), LP(i, 5),
                                                           qkvbL + (size_t)i * QKVN);
  }
  T(g_qw, wg, DMODEL, DMODEL);
  T(g_kw, wg + (size_t)DMODEL * DMODEL, DMODEL, DMODEL);
  T(g_vw, wg + (size_t)2 * DMODEL * DMODEL, DMODEL, DMODEL);
  concat3_kernel<<<(QKVN + 255) / 256, 256, 0, stream>>>(g_qb, g_kb, g_vb, gqkvb);

  // embedding + LN -> hidden (fp32) and xb (bf16)
  embed_kernel<<<NROWS, 128, 0, stream>>>(ids, wemb, pemb, embg, embb, hidden, xb);

  for (int i = 0; i < 4; i++) {
    if (i > 0) {
      gather_cls_kernel<<<(AN * DMODEL + 255) / 256, 256, 0, stream>>>(hidden, xg);
      gemm_bf16<0><<<dim3(AN / 128, QKVN / 128), 256, 0, stream>>>(
          xg, wg, gqkvb, qkvg, nullptr, AN, QKVN, DMODEL);
      graph_attn_kernel<<<dim3(16, NHEAD), 32, 0, stream>>>(qkvg, nodea, hidden, xb);
    }
    // QKV projection
    gemm_bf16<0><<<dim3(NROWS / 128, QKVN / 128), 256, 0, stream>>>(
        xb, wqkv + (size_t)i * QKVN * DMODEL, qkvbL + (size_t)i * QKVN,
        qkv, nullptr, NROWS, QKVN, DMODEL);
    // fused attention
    attn_kernel<<<dim3(AN, NHEAD), 256, 0, stream>>>(qkv, relb, peak, ctx);
    // out-proj + residual (in-place on hidden), then LN1
    gemm_bf16<2><<<dim3(NROWS / 128, DMODEL / 128), 256, 0, stream>>>(
        ctx, wo + (size_t)i * DMODEL * DMODEL, LP(i, 7),
        nullptr, hidden, NROWS, DMODEL, DMODEL);
    ln_kernel<<<NROWS, 128, 0, stream>>>(hidden, LP(i, 8), LP(i, 9), xb);
    // FFN1 (gelu) -> hbuf bf16
    gemm_bf16<1><<<dim3(NROWS / 128, FFDIM / 128), 256, 0, stream>>>(
        xb, wf + (size_t)i * FFDIM * DMODEL, LP(i, 11),
        hbuf, nullptr, NROWS, FFDIM, DMODEL);
    // FFN2 + residual, then LN2 -> hidden / xb
    gemm_bf16<2><<<dim3(NROWS / 128, DMODEL / 128), 256, 0, stream>>>(
        hbuf, wf2 + (size_t)i * DMODEL * FFDIM, LP(i, 13),
        nullptr, hidden, NROWS, DMODEL, FFDIM);
    ln_kernel<<<NROWS, 128, 0, stream>>>(hidden, LP(i, 14), LP(i, 15), xb);
  }
}
